// MegalodonAttention_71846212927933
// MI455X (gfx1250) — matrix operand order repaired and checked by reading, not compiled, not run on hardware
//
#include <hip/hip_runtime.h>
#include <cstdint>
#include <cstddef>

// Problem constants (match reference)
#define BB   2
#define TT   4096
#define HH   16
#define DHH  128
#define DVV  128
#define CCH  1024
#define NCH  4

typedef __attribute__((ext_vector_type(16))) __bf16 v16bf;
typedef __attribute__((ext_vector_type(8)))  float  v8f;

union BF16x16 { v16bf v; uint4 u[2]; };

__device__ __forceinline__ unsigned short f2bf(float f) {
    // round-to-nearest-even float -> bf16 bits
    unsigned int u = __float_as_uint(f);
    unsigned int r = u + 0x7FFFu + ((u >> 16) & 1u);
    return (unsigned short)(r >> 16);
}

__device__ __forceinline__ v16bf ld16(const unsigned short* p) {
    BF16x16 r;
    r.u[0] = *(const uint4*)(p);
    r.u[1] = *(const uint4*)(p + 8);
    return r.v;
}
__device__ __forceinline__ v16bf ld8x2(const unsigned short* plo, const unsigned short* phi) {
    BF16x16 r;
    r.u[0] = *(const uint4*)(plo);
    r.u[1] = *(const uint4*)(phi);
    return r.v;
}

// ---- CDNA5 async global->LDS copy (16B per lane), tracked by ASYNCcnt ----
// VDST = per-lane LDS byte offset VGPR, VADDR = per-lane 64-bit global address.
__device__ __forceinline__ void async_g2lds_b128(const void* g, unsigned ldsoff) {
    unsigned long long ga = (unsigned long long)(uintptr_t)g;
    asm volatile("global_load_async_to_lds_b128 %0, %1, off"
                 :: "v"(ldsoff), "v"(ga) : "memory");
}

// ---- DPP16 xor-style reductions over each 16-lane half (VALU pipe, no LDS) ----
// quad_perm(1,0,3,2)=0xB1 (xor1), quad_perm(2,3,0,1)=0x4E (xor2),
// ROW_HALF_MIRROR=0x141 (xor4-equiv once quads are uniform), ROW_MIRROR=0x140 (xor8-equiv)
__device__ __forceinline__ float dpp_max16(float x) {
    int t;
    t = __builtin_amdgcn_update_dpp(0, __float_as_int(x), 0xB1, 0xF, 0xF, true);
    x = fmaxf(x, __int_as_float(t));
    t = __builtin_amdgcn_update_dpp(0, __float_as_int(x), 0x4E, 0xF, 0xF, true);
    x = fmaxf(x, __int_as_float(t));
    t = __builtin_amdgcn_update_dpp(0, __float_as_int(x), 0x141, 0xF, 0xF, true);
    x = fmaxf(x, __int_as_float(t));
    t = __builtin_amdgcn_update_dpp(0, __float_as_int(x), 0x140, 0xF, 0xF, true);
    x = fmaxf(x, __int_as_float(t));
    return x;
}
__device__ __forceinline__ float dpp_sum16(float x) {
    int t;
    t = __builtin_amdgcn_update_dpp(0, __float_as_int(x), 0xB1, 0xF, 0xF, true);
    x += __int_as_float(t);
    t = __builtin_amdgcn_update_dpp(0, __float_as_int(x), 0x4E, 0xF, 0xF, true);
    x += __int_as_float(t);
    t = __builtin_amdgcn_update_dpp(0, __float_as_int(x), 0x141, 0xF, 0xF, true);
    x += __int_as_float(t);
    t = __builtin_amdgcn_update_dpp(0, __float_as_int(x), 0x140, 0xF, 0xF, true);
    x += __int_as_float(t);
    return x;
}

// ---------------------------------------------------------------------------
// Kernel 1: RoPE(q,k) + bf16 downconvert of q,k,v into head-major [B,H,T,D]
// ---------------------------------------------------------------------------
__global__ void rope_cast_kernel(const float* __restrict__ q, const float* __restrict__ k,
                                 const float* __restrict__ v, const int* __restrict__ start,
                                 unsigned short* __restrict__ wsQ,
                                 unsigned short* __restrict__ wsK,
                                 unsigned short* __restrict__ wsV) {
    int idx = blockIdx.x * blockDim.x + threadIdx.x;
    if (idx >= BB * TT * HH * 64) return;
    int i = idx & 63;            // rotary pair index, half = 64
    int h = (idx >> 6) & 15;
    int t = (idx >> 10) & 4095;
    int b = idx >> 22;

    float pos = (float)(start[0] + t);
    float freq = __expf((float)i * -0.14391156517f);   // exp(-i*ln(1e4)/64)
    float sn, cs;
    __sincosf(pos * freq, &sn, &cs);

    size_t ib = ((size_t)((b * TT + t) * HH + h)) * DHH;   // input  [B,T,H,D]
    size_t ob = ((size_t)((b * HH + h) * TT + t)) * DHH;   // output [B,H,T,D]

    float x1 = q[ib + i], x2 = q[ib + i + 64];
    wsQ[ob + i]      = f2bf(x1 * cs - x2 * sn);
    wsQ[ob + i + 64] = f2bf(x2 * cs + x1 * sn);

    x1 = k[ib + i];  x2 = k[ib + i + 64];
    wsK[ob + i]      = f2bf(x1 * cs - x2 * sn);
    wsK[ob + i + 64] = f2bf(x2 * cs + x1 * sn);

    wsV[ob + i]      = f2bf(v[ib + i]);
    wsV[ob + i + 64] = f2bf(v[ib + i + 64]);
}

// ---------------------------------------------------------------------------
// Kernel 2: chunk-local causal flash attention, bf16 WMMA + fp32 accumulate
// Block = 256 threads = 8 waves; wave owns 16 query rows; block owns 128 rows.
// 64 keys per iteration: 16 S-WMMAs + 16 PV-WMMAs per wave per iteration.
// K staged via global_load_async_to_lds_b128 (ASYNCcnt); V staged transposed.
// ---------------------------------------------------------------------------
#define KBLK 64
#define KP   136   // LDS pitch for K rows   (64 x 128 bf16, +8 pad)
#define VP   72    // LDS pitch for Vt rows  (128 x 64 bf16, +8 pad)
#define PP   72    // LDS pitch for P rows   (16 x 64 bf16 per wave, +8 pad)

__global__ __launch_bounds__(256, 1) void attn_kernel(const unsigned short* __restrict__ wsQ,
                                                      const unsigned short* __restrict__ wsK,
                                                      const unsigned short* __restrict__ wsV,
                                                      float* __restrict__ out) {
    __shared__ __align__(16) unsigned short ldsK[KBLK * KP];      // K block,  row-major [key][dh]
    __shared__ __align__(16) unsigned short ldsVt[128 * VP];      // V block transposed [dv][key]
    __shared__ __align__(16) unsigned short ldsP[8 * 16 * PP];    // per-wave P tiles   [row][key]

    const int qt = blockIdx.x & 7;           // 128-row q tile in chunk
    const int c  = (blockIdx.x >> 3) & 3;    // chunk
    const int h  = (blockIdx.x >> 5) & 15;   // head
    const int b  = blockIdx.x >> 9;          // batch

    const int tid    = threadIdx.x;
    const int wave   = tid >> 5;
    const int lane   = tid & 31;
    const int lo16   = lane & 15;
    const int hihalf = lane >> 4;            // 0 / 1
    const int mbase  = 8 * hihalf;           // C/D rows 0-7 or 8-15

    const size_t headBase = ((size_t)(b * HH + h) * TT + (size_t)c * CCH) * DHH;
    const int q0     = qt * 128 + wave * 16;     // this wave's first q row (in chunk)
    const int wgKend = qt * 128 + 127;           // last key any wave in block needs

    // ---- Q A-fragments (16x32 bf16 per DH chunk), loaded once from global ----
    const unsigned short* Qg = wsQ + headBase + (size_t)(q0 + lo16) * DHH + hihalf * 8;
    v16bf aQ[4];
#pragma unroll
    for (int kk = 0; kk < 4; ++kk)
        aQ[kk] = ld8x2(Qg + kk * 32, Qg + kk * 32 + 16);

    // ---- online-softmax state (rows j=0..7 per lane, per C/D layout) ----
    v8f acc[8];
#pragma unroll
    for (int sb = 0; sb < 8; ++sb) acc[sb] = {};
    float m[8], l[8];
#pragma unroll
    for (int j = 0; j < 8; ++j) { m[j] = -1e30f; l[j] = 0.0f; }

    const int srow = tid >> 2;          // staging: key row 0..63
    const int scol = (tid & 3) * 32;    // staging: 32-elem column group
    const unsigned ldsKoff = (unsigned)(uintptr_t)&ldsK[srow * KP + scol];
    // log2(e)/sqrt(DH): S' = S*SC2, softmax via exp2
    const float SC2 = 0.12751744767f;

    for (int kb = 0; kb <= wgKend; kb += KBLK) {
        // ---- K staging: async global -> LDS, no VGPR round trip ----
        const unsigned short* Kg = wsK + headBase + (size_t)(kb + srow) * DHH + scol;
        async_g2lds_b128(Kg,      ldsKoff);
        async_g2lds_b128(Kg + 8,  ldsKoff + 16);
        async_g2lds_b128(Kg + 16, ldsKoff + 32);
        async_g2lds_b128(Kg + 24, ldsKoff + 48);

        // ---- V staging: load + transpose into LDS [dv][key] ----
        const unsigned short* Vg = wsV + headBase + (size_t)(kb + srow) * DVV + scol;
        unsigned short vloc[32];
        *(uint4*)&vloc[0]  = *(const uint4*)(Vg);
        *(uint4*)&vloc[8]  = *(const uint4*)(Vg + 8);
        *(uint4*)&vloc[16] = *(const uint4*)(Vg + 16);
        *(uint4*)&vloc[24] = *(const uint4*)(Vg + 24);
#pragma unroll
        for (int e = 0; e < 32; ++e)
            ldsVt[(scol + e) * VP + srow] = vloc[e];

        if (kb + KBLK <= wgKend) {                  // prefetch next key block
            __builtin_prefetch(Kg + KBLK * DHH, 0, 1);
            __builtin_prefetch(Vg + KBLK * DVV, 0, 1);
        }
        asm volatile("s_wait_asynccnt 0" ::: "memory");   // drain async K copies
        __syncthreads();

        if (kb <= q0 + 15) {   // this wave still has unmasked keys in the block
            // ---- S = Q . K^T : four 16x16 tiles (keys kb+16t .. kb+16t+15) ----
            v8f s[4];
#pragma unroll
            for (int t = 0; t < 4; ++t) {
                s[t] = {};
#pragma unroll
                for (int kk = 0; kk < 4; ++kk) {
                    v16bf bK = ld8x2(&ldsK[(t * 16 + lo16) * KP + kk * 32 + hihalf * 8], &ldsK[(t * 16 + lo16) * KP + kk * 32 + 16 + hihalf * 8]);
                    s[t] = __builtin_amdgcn_wmma_f32_16x16x32_bf16(false, aQ[kk], false, bK,
                                                                   (short)0, s[t], false, false);
                }
            }

            // ---- scale, causal mask, online softmax (DPP reductions) ----
            unsigned short* pW = &ldsP[wave * 16 * PP];
#pragma unroll
            for (int j = 0; j < 8; ++j) {
                const int qg = q0 + mbase + j;
                float a[4];
#pragma unroll
                for (int t = 0; t < 4; ++t) {
                    float av = s[t][j] * SC2;
                    a[t] = (kb + t * 16 + lo16 <= qg) ? av : -1e30f;
                }
                float rmax = fmaxf(fmaxf(a[0], a[1]), fmaxf(a[2], a[3]));
                rmax = dpp_max16(rmax);

                float mn    = fmaxf(m[j], rmax);
                float alpha = exp2f(m[j] - mn);
                m[j] = mn;

                float p0 = exp2f(a[0] - mn);
                float p1 = exp2f(a[1] - mn);
                float p2 = exp2f(a[2] - mn);
                float p3 = exp2f(a[3] - mn);
                pW[(mbase + j) * PP + lo16]      = f2bf(p0);
                pW[(mbase + j) * PP + 16 + lo16] = f2bf(p1);
                pW[(mbase + j) * PP + 32 + lo16] = f2bf(p2);
                pW[(mbase + j) * PP + 48 + lo16] = f2bf(p3);

                float rs = (p0 + p1) + (p2 + p3);
                rs = dpp_sum16(rs);
                l[j] = l[j] * alpha + rs;
#pragma unroll
                for (int sb = 0; sb < 8; ++sb) acc[sb][j] *= alpha;
            }
            asm volatile("s_wait_dscnt 0" ::: "memory");   // wave-local LDS transpose

            // ---- P A-fragments (two 16x32 halves of the 16x64 P tile) ----
            const unsigned short* pb = &ldsP[wave * 16 * PP + lo16 * PP + hihalf * 8];
            v16bf pA0 = ld8x2(pb,      pb + 16);
            v16bf pA1 = ld8x2(pb + 32, pb + 48);

            // ---- O += P . V : 8 DV slabs x 2 key-chunks of 32 ----
#pragma unroll
            for (int sb = 0; sb < 8; ++sb) {
                v16bf bV0 = ld8x2(&ldsVt[(sb * 16 + lo16) * VP + hihalf * 8], &ldsVt[(sb * 16 + lo16) * VP + 16 + hihalf * 8]);
                acc[sb] = __builtin_amdgcn_wmma_f32_16x16x32_bf16(false, pA0, false, bV0,
                                                                  (short)0, acc[sb], false, false);
                v16bf bV1 = ld8x2(&ldsVt[(sb * 16 + lo16) * VP + 32 + hihalf * 8], &ldsVt[(sb * 16 + lo16) * VP + 48 + hihalf * 8]);
                acc[sb] = __builtin_amdgcn_wmma_f32_16x16x32_bf16(false, pA1, false, bV1,
                                                                  (short)0, acc[sb], false, false);
            }
        }
        __syncthreads();   // before next block overwrites ldsK/ldsVt
    }

    // ---- normalize and store: out[b][c*CCH+row][h][dv], fp32 ----
    const size_t outBase = ((size_t)b * TT + (size_t)c * CCH) * (HH * DVV) + (size_t)h * DVV;
#pragma unroll
    for (int sb = 0; sb < 8; ++sb) {
#pragma unroll
        for (int j = 0; j < 8; ++j) {
            int row = q0 + mbase + j;
            int dv  = sb * 16 + lo16;
            out[outBase + (size_t)row * (HH * DVV) + dv] = acc[sb][j] / l[j];
        }
    }
}

// ---------------------------------------------------------------------------
extern "C" void kernel_launch(void* const* d_in, const int* in_sizes, int n_in,
                              void* d_out, int out_size, void* d_ws, size_t ws_size,
                              hipStream_t stream) {
    const float* q     = (const float*)d_in[0];
    const float* k     = (const float*)d_in[1];
    const float* v     = (const float*)d_in[2];
    const int*   start = (const int*)d_in[3];
    float*       out   = (float*)d_out;

    const size_t perTensor = (size_t)BB * HH * TT * DHH;   // 16,777,216 bf16 elems
    unsigned short* ws  = (unsigned short*)d_ws;
    unsigned short* wsQ = ws;
    unsigned short* wsK = ws + perTensor;
    unsigned short* wsV = ws + 2 * perTensor;

    int npairs = BB * TT * HH * 64;
    rope_cast_kernel<<<(npairs + 255) / 256, 256, 0, stream>>>(q, k, v, start, wsQ, wsK, wsV);

    // one block per (b, h, chunk, 128-row q tile) = 2*16*4*8 = 1024 blocks
    attn_kernel<<<BB * HH * NCH * 8, 256, 0, stream>>>(wsQ, wsK, wsV, out);
}
